// TopkGate_37022618091811
// MI455X (gfx1250) — compile-verified, matrix operand order checked by reading, not run on hardware
//
#include <hip/hip_runtime.h>
#include <hip/hip_bf16.h>

#define T_TOKENS 16384
#define D_DIM    2048
#define E_EXP    64
#define TOPK     2
#define BLOCK_T  128
#define NBLOCKS  (T_TOKENS / BLOCK_T)   // 128
#define CHUNK_D  256
#define NCHUNKS  (D_DIM / CHUNK_D)      // 8
#define WG_Q4    33                     // uint4 per expert row: 32 data + 1 pad -> 528B stride
#define LPAD     65

typedef __attribute__((ext_vector_type(16))) __bf16       v16bf;
typedef __attribute__((ext_vector_type(8)))  float        v8f;
typedef __attribute__((ext_vector_type(8)))  unsigned int v8u;

__device__ __forceinline__ unsigned pk_bf16(float lo, float hi) {
#if __has_builtin(__builtin_amdgcn_cvt_pk_bf16_f32)
    return __builtin_bit_cast(unsigned, __builtin_amdgcn_cvt_pk_bf16_f32(lo, hi));
#else
    // pack two fp32 -> two bf16 (round-half-up), low element in [15:0]
    unsigned a = __float_as_uint(lo) + 0x8000u;
    unsigned b = __float_as_uint(hi) + 0x8000u;
    return (a >> 16) | (b & 0xFFFF0000u);
#endif
}

// ---------------------------------------------------------------------------
// Pass 1: bf16-WMMA logits GEMM + softmax + entropy + top-2 + block partials
// ---------------------------------------------------------------------------
__global__ __launch_bounds__(256)
void moe_gate_main(const float* __restrict__ x, const float* __restrict__ wg,
                   float* __restrict__ out, int* __restrict__ blkCounts,
                   float* __restrict__ blkMe, float* __restrict__ blkEnt)
{
    __shared__ union {
        uint4 wgb4[E_EXP][WG_Q4];                    // 33792 B bf16 staging, 528B row stride
        struct {
            float logits[BLOCK_T][LPAD];             // 33280 B
            float mrow[BLOCK_T];
            float inv[BLOCK_T];
            float entr[BLOCK_T];
            int   i0[BLOCK_T];
            int   i1[BLOCK_T];
        } s;
    } u;

    const int tid  = threadIdx.x;
    const int wave = tid >> 5;
    const int lane = tid & 31;
    const int half = lane >> 4;     // K-subgroup select for 16-bit A/B layout
    const int l16  = lane & 15;
    const int blk  = blockIdx.x;

    const int rowTok = blk * BLOCK_T + wave * 16 + l16;
    const float* xrow = x + (size_t)rowTok * D_DIM;

    v8f acc[4] = {};   // 4 N-tiles (64 experts) x 16x16 f32

    for (int c = 0; c < NCHUNKS; ++c) {
        const int cb = c * CHUNK_D;
        __syncthreads();
        // stage wg chunk fp32 -> bf16 into padded LDS (coalesced float4 loads)
        #pragma unroll
        for (int j = 0; j < (E_EXP * CHUNK_D / 4) / 256; ++j) {  // 16 iters
            int i = j * 256 + tid;           // float4 slot
            int r = i >> 6;                  // 64 float4 slots per expert row
            int s = i & 63;                  // slot within row (8 bytes bf16)
            float4 f = *(const float4*)(wg + (size_t)r * D_DIM + cb + s * 4);
            ((uint2*)u.wgb4[r])[s] = make_uint2(pk_bf16(f.x, f.y), pk_bf16(f.z, f.w));
        }
        __syncthreads();

        if (c + 1 < NCHUNKS)  // global_prefetch_b8 for next x chunk
            __builtin_prefetch(xrow + cb + CHUNK_D + half * 128, 0, 3);

        for (int kc = 0; kc < CHUNK_D; kc += 32) {
            // A fragment: 16 bf16 per lane, K groups [half*8..+8) and [16+half*8..+8)
            const int col0 = cb + kc + half * 8;
            float4 a0 = *(const float4*)(xrow + col0);
            float4 a1 = *(const float4*)(xrow + col0 + 4);
            float4 a2 = *(const float4*)(xrow + col0 + 16);
            float4 a3 = *(const float4*)(xrow + col0 + 20);
            v8u au;
            au[0] = pk_bf16(a0.x, a0.y); au[1] = pk_bf16(a0.z, a0.w);
            au[2] = pk_bf16(a1.x, a1.y); au[3] = pk_bf16(a1.z, a1.w);
            au[4] = pk_bf16(a2.x, a2.y); au[5] = pk_bf16(a2.z, a2.w);
            au[6] = pk_bf16(a3.x, a3.y); au[7] = pk_bf16(a3.z, a3.w);
            v16bf aF = __builtin_bit_cast(v16bf, au);

            const int q = (kc >> 3) + half;  // uint4 index of K-group 0 in row

            #pragma unroll
            for (int n = 0; n < 4; ++n) {
                const int e = n * 16 + l16;   // B lane = expert column (wg row-major)
                uint4 b0 = u.wgb4[e][q];      // ds_load_b128
                uint4 b1 = u.wgb4[e][q + 2];  // ds_load_b128
                v8u bu;
                bu[0] = b0.x; bu[1] = b0.y; bu[2] = b0.z; bu[3] = b0.w;
                bu[4] = b1.x; bu[5] = b1.y; bu[6] = b1.z; bu[7] = b1.w;
                v16bf bF = __builtin_bit_cast(v16bf, bu);
                acc[n] = __builtin_amdgcn_wmma_f32_16x16x32_bf16(
                    false, aF, false, bF, (short)0, acc[n], false, false);
            }
        }
    }

    __syncthreads();
    // C layout: VGPR r, lanes 0-15 -> M=r, lanes 16-31 -> M=8+r; N = lane%16 + 16n
    #pragma unroll
    for (int n = 0; n < 4; ++n)
        #pragma unroll
        for (int r = 0; r < 8; ++r)
            u.s.logits[wave * 16 + half * 8 + r][n * 16 + l16] = acc[n][r];
    __syncthreads();

    // per-token softmax / entropy / stable top-2
    if (tid < BLOCK_T) {
        const float* L = u.s.logits[tid];
        float mx = L[0];
        for (int e = 1; e < E_EXP; ++e) mx = fmaxf(mx, L[e]);
        float s1 = 0.f, s2 = 0.f;
        float v0 = -3.4e38f, v1 = -3.4e38f; int b0 = 0, b1 = 1;
        for (int e = 0; e < E_EXP; ++e) {
            float d  = L[e] - mx;
            float ex = __expf(d);
            s1 += ex; s2 += ex * d;
            if (L[e] > v0)      { v1 = v0; b1 = b0; v0 = L[e]; b0 = e; }
            else if (L[e] > v1) { v1 = L[e]; b1 = e; }
        }
        float inv = 1.0f / s1;
        float ent = logf(s1) - s2 * inv;          // -sum(p*logp)
        int tg = blk * BLOCK_T + tid;
        float* idxOut  = out + 2;
        float* gateOut = out + 2 + 2 * T_TOKENS * TOPK;
        idxOut[tg * 2 + 0]  = (float)b0;
        idxOut[tg * 2 + 1]  = (float)b1;
        gateOut[tg * 2 + 0] = __expf(v0 - mx) * inv;
        gateOut[tg * 2 + 1] = __expf(v1 - mx) * inv;
        u.s.mrow[tid] = mx; u.s.inv[tid] = inv; u.s.entr[tid] = ent;
        u.s.i0[tid] = b0;   u.s.i1[tid] = b1;
    }
    __syncthreads();

    // block partials: per-expert gate sums (me), counts per (k,e), entropy sum
    if (tid < 64) {
        int e = tid;
        float s = 0.f;
        for (int t = 0; t < BLOCK_T; ++t)
            s += __expf(u.s.logits[t][e] - u.s.mrow[t]) * u.s.inv[t];
        blkMe[blk * 64 + e] = s;
    } else if (tid == 64) {
        float s = 0.f;
        for (int t = 0; t < BLOCK_T; ++t) s += u.s.entr[t];
        blkEnt[blk] = s;
    } else if (tid >= 128) {
        int k = (tid - 128) >> 6;
        int e = (tid - 128) & 63;
        const int* ip = k ? u.s.i1 : u.s.i0;
        int cnt = 0;
        for (int t = 0; t < BLOCK_T; ++t) cnt += (ip[t] == e) ? 1 : 0;
        blkCounts[(blk * 2 + k) * 64 + e] = cnt;
    }
}

// ---------------------------------------------------------------------------
// Pass 2: deterministic exclusive scan over blocks + scalars (l_aux, entropy)
// ---------------------------------------------------------------------------
__global__ __launch_bounds__(128)
void moe_gate_scan(const int* __restrict__ blkCounts, const float* __restrict__ blkMe,
                   const float* __restrict__ blkEnt, int* __restrict__ blkOffs,
                   float* __restrict__ out)
{
    __shared__ int   total0[64];
    __shared__ float prod[64];
    int tid = threadIdx.x;
    int k = tid >> 6, e = tid & 63;
    int run = 0;
    for (int b = 0; b < NBLOCKS; ++b) {
        int idx = (b * 2 + k) * 64 + e;
        blkOffs[idx] = run;
        run += blkCounts[idx];
    }
    if (k == 0) total0[e] = run;        // counts[k=0][e] for ce + acc_base
    __syncthreads();
    if (k == 1) {
        int t0 = total0[e];             // acc_base for slot 1
        for (int b = 0; b < NBLOCKS; ++b)
            blkOffs[(b * 2 + 1) * 64 + e] += t0;
    } else {
        float me = 0.f;
        for (int b = 0; b < NBLOCKS; ++b) me += blkMe[b * 64 + e];
        float ce = (float)run * ((float)E_EXP / (float)T_TOKENS);
        prod[e] = me * ce;
    }
    __syncthreads();
    if (tid == 0) {
        float la = 0.f;
        for (int i = 0; i < 64; ++i) la += prod[i];
        out[0] = la / (float)T_TOKENS;
        float en = 0.f;
        for (int b = 0; b < NBLOCKS; ++b) en += blkEnt[b];
        out[1] = en / (float)T_TOKENS;
    }
}

// ---------------------------------------------------------------------------
// Pass 3: token-ordered capacity locations per (k,expert)
// ---------------------------------------------------------------------------
__global__ __launch_bounds__(128)
void moe_gate_locs(const float* __restrict__ outIdx, const int* __restrict__ blkOffs,
                   float* __restrict__ outLoc)
{
    __shared__ int sidx[BLOCK_T * 2];
    int tid = threadIdx.x, blk = blockIdx.x;
    for (int i = tid; i < BLOCK_T * 2; i += 128)
        sidx[i] = (int)outIdx[blk * BLOCK_T * 2 + i];
    __syncthreads();
    int k = tid >> 6, e = tid & 63;
    int run = blkOffs[(blk * 2 + k) * 64 + e];
    for (int t = 0; t < BLOCK_T; ++t)
        if (sidx[t * 2 + k] == e)
            outLoc[(blk * BLOCK_T + t) * 2 + k] = (float)(run++);
}

// ---------------------------------------------------------------------------
extern "C" void kernel_launch(void* const* d_in, const int* in_sizes, int n_in,
                              void* d_out, int out_size, void* d_ws, size_t ws_size,
                              hipStream_t stream)
{
    const float* x  = (const float*)d_in[0];   // [T, D] fp32
    const float* wg = (const float*)d_in[1];   // [E, D] fp32
    float* out = (float*)d_out;                // [laux, ent, idx(T*2), loc(T*2), gates(T*2)]
    char*  ws  = (char*)d_ws;

    int*   blkCounts = (int*)(ws);             // 128*2*64 ints  = 64 KB
    int*   blkOffs   = (int*)(ws + 65536);     // 64 KB
    float* blkMe     = (float*)(ws + 131072);  // 32 KB
    float* blkEnt    = (float*)(ws + 163840);  // 512 B

    moe_gate_main<<<NBLOCKS, 256, 0, stream>>>(x, wg, out, blkCounts, blkMe, blkEnt);
    moe_gate_scan<<<1, 128, 0, stream>>>(blkCounts, blkMe, blkEnt, blkOffs, out);

    float* idxOut = out + 2;
    float* locOut = out + 2 + T_TOKENS * TOPK;
    moe_gate_locs<<<NBLOCKS, 128, 0, stream>>>(idxOut, blkOffs, locOut);
}
